// MSA_31318901522524
// MI455X (gfx1250) — matrix operand order repaired and checked by reading, not compiled, not run on hardware
//
#include <hip/hip_runtime.h>
#include <hip/hip_bf16.h>

// ---------------------------------------------------------------------------
// MI455X (gfx1250) fused MSA:
//   qkv GEMMs (bf16 WMMA, ping-pong double-buffered) ->
//   fused energy/softmax(HEAD-axis)/att@v -> out proj
// All matrix math via v_wmma_f32_16x16x32_bf16 (f32 accumulate).
// Workspace (~168 MB, fits the 192MB L2):
//   xb [16384][1024] bf16 | wqb/wkb/wvb/wpb [1024][1024] bf16
//   qb,kb [B][H][N][D] bf16 | vtb [B][H][D][N] bf16 | ob [B][N][E] bf16
// ---------------------------------------------------------------------------

typedef __attribute__((ext_vector_type(16))) __bf16          v16bf;
typedef __attribute__((ext_vector_type(8)))  float           v8f;
typedef __attribute__((ext_vector_type(8)))  unsigned short  u16x8;

#define B_ 16
#define N_ 1024
#define E_ 1024
#define H_ 16
#define D_ 64
#define M_TOT (B_ * N_)   // 16384

static __device__ __forceinline__ unsigned short f2bf(float f) {
  unsigned int u = __float_as_uint(f);
  u += 0x7FFFu + ((u >> 16) & 1u);   // round-to-nearest-even
  return (unsigned short)(u >> 16);
}

union FragU { v16bf v; u16x8 h[2]; };

// A fragment: tile row-major [16][lda] bf16. Per CDNA5 ISA 16-bit A layout:
// lane row m = lane&15; lanes<16 hold K {0..7,16..23}, lanes>=16 hold K {8..15,24..31}.
static __device__ __forceinline__ v16bf load_frag_a(const unsigned short* p0, int lda, int lane) {
  int m  = lane & 15;
  int kb = (lane < 16) ? 0 : 8;
  const unsigned short* p = p0 + m * lda + kb;
  FragU f;
  f.h[0] = *(const u16x8*)(p);
  f.h[1] = *(const u16x8*)(p + 16);
  return f.v;
}

// B fragment from Bt = N-major [16][ldb] (row n holds K contiguous).
// Per ISA: lane col n = lane&15; lanes<16 hold K {0..7,16..23}, lanes>=16 hold K {8..15,24..31}.
static __device__ __forceinline__ v16bf load_frag_b(const unsigned short* p0, int ldb, int lane) {
  int n  = lane & 15;
  int kb = (lane < 16) ? 0 : 8;
  const unsigned short* p = p0 + n * ldb + kb;
  FragU f;
  f.h[0] = *(const u16x8*)(p);
  f.h[1] = *(const u16x8*)(p + 16);
  return f.v;
}

static __device__ __forceinline__ v8f wmma_bf16(v16bf a, v16bf b, v8f c) {
  return __builtin_amdgcn_wmma_f32_16x16x32_bf16(false, a, false, b, (short)0, c, false, false);
}

// ---------------------------------------------------------------------------
__global__ __launch_bounds__(256)
void k_cvt(const float* __restrict__ src, unsigned int* __restrict__ dst, int npairs) {
  const float2* s2 = (const float2*)src;
  for (int i = blockIdx.x * blockDim.x + threadIdx.x; i < npairs; i += gridDim.x * blockDim.x) {
    float2 f = s2[i];
    dst[i] = (unsigned int)f2bf(f.x) | ((unsigned int)f2bf(f.y) << 16);
  }
}

// ---- shared GEMM core: 32(M) x 64(N) wave tile, ping-pong double buffering.
// Accumulates C = X @ W.T over K = E_ into acc[2][4]; no register copies.
static __device__ __forceinline__ void gemm_core(const unsigned short* __restrict__ arow0,
                                                 const unsigned short* __restrict__ arow1,
                                                 const unsigned short* __restrict__ wrow,
                                                 int lane, v8f acc[2][4]) {
  v16bf a0[2], b0[4], a1[2], b1[4];
  a0[0] = load_frag_a(arow0, E_, lane);
  a0[1] = load_frag_a(arow1, E_, lane);
#pragma unroll
  for (int j = 0; j < 4; ++j) b0[j] = load_frag_b(wrow + j * 16 * E_, E_, lane);

  for (int k0 = 0; k0 < E_; k0 += 64) {
    // load set1 <- k0+32 (always in range: E_ % 64 == 0)
    __builtin_prefetch(arow0 + k0 + 64, 0, 1);            // global_prefetch_b8
    a1[0] = load_frag_a(arow0 + k0 + 32, E_, lane);
    a1[1] = load_frag_a(arow1 + k0 + 32, E_, lane);
#pragma unroll
    for (int j = 0; j < 4; ++j) b1[j] = load_frag_b(wrow + j * 16 * E_ + k0 + 32, E_, lane);

    // compute with set0 (k0)
#pragma unroll
    for (int i = 0; i < 2; ++i)
#pragma unroll
      for (int j = 0; j < 4; ++j)
        acc[i][j] = wmma_bf16(a0[i], b0[j], acc[i][j]);

    // load set0 <- k0+64 (guarded: must not read past the buffer)
    if (k0 + 64 < E_) {
      a0[0] = load_frag_a(arow0 + k0 + 64, E_, lane);
      a0[1] = load_frag_a(arow1 + k0 + 64, E_, lane);
#pragma unroll
      for (int j = 0; j < 4; ++j) b0[j] = load_frag_b(wrow + j * 16 * E_ + k0 + 64, E_, lane);
    }

    // compute with set1 (k0+32)
#pragma unroll
    for (int i = 0; i < 2; ++i)
#pragma unroll
      for (int j = 0; j < 4; ++j)
        acc[i][j] = wmma_bf16(a1[i], b1[j], acc[i][j]);
  }
}

// out[m,n] = sum_k X[m,k]*W[n,k] + bias[n]; bf16 store to head-split layout.
// vtranspose==0 : dst[((b*H+h)*N + npos)*D + d]     (q, k)
// vtranspose==1 : dst[((b*H+h)*D + d)*N + npos]     (v, pre-transposed)
__global__ __launch_bounds__(256)
void k_qkv_gemm(const unsigned short* __restrict__ Xb,
                const unsigned short* __restrict__ Wb,
                const float* __restrict__ bias,
                unsigned short* __restrict__ dst,
                int vtranspose) {
  int lane = threadIdx.x & 31;
  int wave = blockIdx.x * (blockDim.x >> 5) + (threadIdx.x >> 5);
  int mt = wave >> 4;      // 512 row-tiles of 32
  int nt = wave & 15;      // 16 col-tiles of 64
  const unsigned short* arow0 = Xb + (mt * 32) * E_;

  v8f acc[2][4] = {};
  gemm_core(arow0, arow0 + 16 * E_, Wb + (nt * 64) * E_, lane, acc);

  int mhi = (lane >> 4) * 8, nloc = lane & 15;
#pragma unroll
  for (int i = 0; i < 2; ++i)
#pragma unroll
    for (int j = 0; j < 4; ++j)
#pragma unroll
      for (int r = 0; r < 8; ++r) {
        int m = mt * 32 + i * 16 + mhi + r;
        int n = nt * 64 + j * 16 + nloc;
        unsigned short bf = f2bf(acc[i][j][r] + bias[n]);
        int bb = m >> 10, npos = m & (N_ - 1);
        int hh = n >> 6,  d    = n & (D_ - 1);
        int idx = vtranspose ? (((bb * H_ + hh) * D_ + d) * N_ + npos)
                             : (((bb * H_ + hh) * N_ + npos) * D_ + d);
        dst[idx] = bf;
      }
}

// Fused: energy (per head) -> softmax over HEAD axis -> /32 -> att@v
__global__ __launch_bounds__(512)
void k_attn(const unsigned short* __restrict__ Qb,
            const unsigned short* __restrict__ Kb,
            const unsigned short* __restrict__ VTb,
            unsigned short* __restrict__ Ob) {
  __shared__ __align__(16) float          e_lds[H_][2][16][16];   // 32 KB
  __shared__ __align__(16) float          mx_lds[2][16][16];      //  2 KB
  __shared__ __align__(16) float          rs_lds[2][16][16];      //  2 KB
  __shared__ __align__(16) unsigned short att_lds[H_][16][32];    // 16 KB

  int lane = threadIdx.x & 31;
  int h    = threadIdx.x >> 5;          // wave == head
  int b    = blockIdx.x >> 6;
  int qt   = blockIdx.x & 63;

  const unsigned short* qtile  = Qb  + ((b * H_ + h) * N_ + qt * 16) * D_;
  const unsigned short* kmat   = Kb  + ((h * H_ + b) * N_) * D_;   // 'hbkd' swap (B==H)
  const unsigned short* vthead = VTb + ((b * H_ + h) * D_) * N_;

  int mhi = (lane >> 4) * 8, nloc = lane & 15;
  v8f oacc[4] = {};

  // q A-fragments are k-loop invariant: load once.
  v16bf aq[2];
#pragma unroll
  for (int d0 = 0; d0 < 2; ++d0) aq[d0] = load_frag_a(qtile + d0 * 32, D_, lane);

  for (int kt = 0; kt < N_ / 32; ++kt) {
    // v B-fragments issued early: latency hides behind energy WMMAs + softmax.
    v16bf vfrag[4];
#pragma unroll
    for (int dt = 0; dt < 4; ++dt)
      vfrag[dt] = load_frag_b(vthead + (dt * 16) * N_ + kt * 32, N_, lane);

    // energy: two 16x16 subtiles over kpos, K-dim = D (two x32 steps)
    v8f e[2] = {};
#pragma unroll
    for (int d0 = 0; d0 < 2; ++d0) {
#pragma unroll
      for (int s = 0; s < 2; ++s) {
        v16bf kf = load_frag_b(kmat + (kt * 32 + s * 16) * D_ + d0 * 32, D_, lane);
        e[s] = wmma_bf16(aq[d0], kf, e[s]);
      }
    }
#pragma unroll
    for (int s = 0; s < 2; ++s)
#pragma unroll
      for (int r = 0; r < 8; ++r)
        e_lds[h][s][mhi + r][nloc] = e[s][r];
    __syncthreads();

    // softmax over HEAD axis: one (s,m,n) element per thread (512 == 2*256)
    {
      int t = threadIdx.x;
      int s = t >> 8, idx = t & 255, m = idx >> 4, n = idx & 15;
      float mx = -1e30f;
#pragma unroll
      for (int j = 0; j < H_; ++j) mx = fmaxf(mx, e_lds[j][s][m][n]);
      float sm = 0.0f;
#pragma unroll
      for (int j = 0; j < H_; ++j) sm += __expf(e_lds[j][s][m][n] - mx);
      mx_lds[s][m][n] = mx;
      rs_lds[s][m][n] = 1.0f / (sm * 32.0f);   // scale = sqrt(E) applied AFTER softmax
    }
    __syncthreads();

#pragma unroll
    for (int s = 0; s < 2; ++s)
#pragma unroll
      for (int r = 0; r < 8; ++r) {
        int m = mhi + r;
        float av = __expf(e[s][r] - mx_lds[s][m][nloc]) * rs_lds[s][m][nloc];
        att_lds[h][m][s * 16 + nloc] = f2bf(av);
      }
    __syncthreads();

    // o += att[16x32] @ v[32x64]
    v16bf afrag = load_frag_a(&att_lds[h][0][0], 32, lane);
#pragma unroll
    for (int dt = 0; dt < 4; ++dt)
      oacc[dt] = wmma_bf16(afrag, vfrag[dt], oacc[dt]);
  }

  // write o merged back to [B][N][E]
#pragma unroll
  for (int dt = 0; dt < 4; ++dt)
#pragma unroll
    for (int r = 0; r < 8; ++r) {
      int m   = qt * 16 + mhi + r;
      int col = h * D_ + dt * 16 + nloc;
      Ob[(b * N_ + m) * E_ + col] = f2bf(oacc[dt][r]);
    }
}

// out = o @ Wp.T + bp  (fp32 output), same ping-pong 32x64 wave tile
__global__ __launch_bounds__(256)
void k_proj(const unsigned short* __restrict__ Ab,
            const unsigned short* __restrict__ Wb,
            const float* __restrict__ bias,
            float* __restrict__ out) {
  int lane = threadIdx.x & 31;
  int wave = blockIdx.x * (blockDim.x >> 5) + (threadIdx.x >> 5);
  int mt = wave >> 4;
  int nt = wave & 15;
  const unsigned short* arow0 = Ab + (mt * 32) * E_;

  v8f acc[2][4] = {};
  gemm_core(arow0, arow0 + 16 * E_, Wb + (nt * 64) * E_, lane, acc);

  int mhi = (lane >> 4) * 8, nloc = lane & 15;
#pragma unroll
  for (int i = 0; i < 2; ++i)
#pragma unroll
    for (int j = 0; j < 4; ++j)
#pragma unroll
      for (int r = 0; r < 8; ++r) {
        int m = mt * 32 + i * 16 + mhi + r;
        int n = nt * 64 + j * 16 + nloc;
        out[m * E_ + n] = acc[i][j][r] + bias[n];
      }
}

// ---------------------------------------------------------------------------
extern "C" void kernel_launch(void* const* d_in, const int* in_sizes, int n_in,
                              void* d_out, int out_size, void* d_ws, size_t ws_size,
                              hipStream_t stream) {
  (void)in_sizes; (void)n_in; (void)out_size; (void)ws_size;
  const float* x  = (const float*)d_in[0];
  const float* Wq = (const float*)d_in[1];
  const float* bq = (const float*)d_in[2];
  const float* Wk = (const float*)d_in[3];
  const float* bk = (const float*)d_in[4];
  const float* Wv = (const float*)d_in[5];
  const float* bv = (const float*)d_in[6];
  const float* Wp = (const float*)d_in[7];
  const float* bp = (const float*)d_in[8];

  char* ws = (char*)d_ws;
  size_t off = 0;
  unsigned short* xb  = (unsigned short*)(ws + off); off += (size_t)M_TOT * E_ * 2;
  unsigned short* wqb = (unsigned short*)(ws + off); off += (size_t)E_ * E_ * 2;
  unsigned short* wkb = (unsigned short*)(ws + off); off += (size_t)E_ * E_ * 2;
  unsigned short* wvb = (unsigned short*)(ws + off); off += (size_t)E_ * E_ * 2;
  unsigned short* wpb = (unsigned short*)(ws + off); off += (size_t)E_ * E_ * 2;
  unsigned short* qb  = (unsigned short*)(ws + off); off += (size_t)M_TOT * E_ * 2;
  unsigned short* kb  = (unsigned short*)(ws + off); off += (size_t)M_TOT * E_ * 2;
  unsigned short* vtb = (unsigned short*)(ws + off); off += (size_t)M_TOT * E_ * 2;
  unsigned short* ob  = (unsigned short*)(ws + off); off += (size_t)M_TOT * E_ * 2;

  // fp32 -> bf16 staging (2 elements per thread-iteration)
  k_cvt<<<2048, 256, 0, stream>>>(x,  (unsigned int*)xb,  M_TOT * E_ / 2);
  k_cvt<<<512,  256, 0, stream>>>(Wq, (unsigned int*)wqb, E_ * E_ / 2);
  k_cvt<<<512,  256, 0, stream>>>(Wk, (unsigned int*)wkb, E_ * E_ / 2);
  k_cvt<<<512,  256, 0, stream>>>(Wv, (unsigned int*)wvb, E_ * E_ / 2);
  k_cvt<<<512,  256, 0, stream>>>(Wp, (unsigned int*)wpb, E_ * E_ / 2);

  // q/k/v projections: 8192 waves each (512 m-tiles x 16 n-tiles), 8 waves/block
  k_qkv_gemm<<<1024, 256, 0, stream>>>(xb, wqb, bq, qb,  0);
  k_qkv_gemm<<<1024, 256, 0, stream>>>(xb, wkb, bk, kb,  0);
  k_qkv_gemm<<<1024, 256, 0, stream>>>(xb, wvb, bv, vtb, 1);

  // fused attention: (b, q-tile) per block, one wave per head
  k_attn<<<B_ * (N_ / 16), 512, 0, stream>>>(qb, kb, vtb, ob);

  // output projection
  k_proj<<<1024, 256, 0, stream>>>(ob, wpb, bp, (float*)d_out);
}
